// Samba_3959959847438
// MI455X (gfx1250) — compile-verified, matrix operand order checked by reading, not run on hardware
//
#include <hip/hip_runtime.h>
#include <hip/hip_bf16.h>
#include <math.h>

// ---------------- model dims ----------------
#define V_  50280
#define DM_ 768
#define NL_ 8
#define NS_ 16
#define E_  1536
#define DR_ 48
#define S_  512
#define SD_ 24576          // E_*NS_
#define HID_ 512

typedef __attribute__((ext_vector_type(16))) __bf16 v16bf;
typedef __attribute__((ext_vector_type(8)))  float  v8f;

__device__ __forceinline__ v8f wmma_bf16(v16bf a, v16bf b, v8f c) {
    return __builtin_amdgcn_wmma_f32_16x16x32_bf16(
        false, a, false, b, (short)0, c, false, false);
}

// Fast fragment load: 4 x global_load_b128 + native f32->bf16 converts.
// Fragment layout for 16-bit A/B operands (wave32):
//   lanes 0-15  hold K = [ka..ka+7] , [kb..kb+7]  of row l16
//   lanes 16-31 hold K = [ka..ka+7] , [kb..kb+7]  (ka/kb already offset by +8)
__device__ __forceinline__ v16bf load_frag(const float* __restrict__ r,
                                           int ka, int kb) {
    float4 a = *reinterpret_cast<const float4*>(r + ka);
    float4 b = *reinterpret_cast<const float4*>(r + ka + 4);
    float4 c = *reinterpret_cast<const float4*>(r + kb);
    float4 d = *reinterpret_cast<const float4*>(r + kb + 4);
    v16bf t;
    t[0] = (__bf16)a.x; t[1] = (__bf16)a.y; t[2] = (__bf16)a.z; t[3] = (__bf16)a.w;
    t[4] = (__bf16)b.x; t[5] = (__bf16)b.y; t[6] = (__bf16)b.z; t[7] = (__bf16)b.w;
    t[8] = (__bf16)c.x; t[9] = (__bf16)c.y; t[10]= (__bf16)c.z; t[11]= (__bf16)c.w;
    t[12]= (__bf16)d.x; t[13]= (__bf16)d.y; t[14]= (__bf16)d.z; t[15]= (__bf16)d.w;
    return t;
}

__device__ __forceinline__ v16bf load_frag_avg(const float* __restrict__ r,
                                               const float* __restrict__ r2,
                                               int ka, int kb) {
    float4 a = *reinterpret_cast<const float4*>(r + ka);
    float4 b = *reinterpret_cast<const float4*>(r + ka + 4);
    float4 c = *reinterpret_cast<const float4*>(r + kb);
    float4 d = *reinterpret_cast<const float4*>(r + kb + 4);
    float4 a2 = *reinterpret_cast<const float4*>(r2 + ka);
    float4 b2 = *reinterpret_cast<const float4*>(r2 + ka + 4);
    float4 c2 = *reinterpret_cast<const float4*>(r2 + kb);
    float4 d2 = *reinterpret_cast<const float4*>(r2 + kb + 4);
    v16bf t;
    t[0] = (__bf16)(0.5f*(a.x+a2.x)); t[1] = (__bf16)(0.5f*(a.y+a2.y));
    t[2] = (__bf16)(0.5f*(a.z+a2.z)); t[3] = (__bf16)(0.5f*(a.w+a2.w));
    t[4] = (__bf16)(0.5f*(b.x+b2.x)); t[5] = (__bf16)(0.5f*(b.y+b2.y));
    t[6] = (__bf16)(0.5f*(b.z+b2.z)); t[7] = (__bf16)(0.5f*(b.w+b2.w));
    t[8] = (__bf16)(0.5f*(c.x+c2.x)); t[9] = (__bf16)(0.5f*(c.y+c2.y));
    t[10]= (__bf16)(0.5f*(c.z+c2.z)); t[11]= (__bf16)(0.5f*(c.w+c2.w));
    t[12]= (__bf16)(0.5f*(d.x+d2.x)); t[13]= (__bf16)(0.5f*(d.y+d2.y));
    t[14]= (__bf16)(0.5f*(d.z+d2.z)); t[15]= (__bf16)(0.5f*(d.w+d2.w));
    return t;
}

// Guarded (zero-padding) loader for the K remainder (only K=48 uses this).
__device__ __forceinline__ v16bf load_frag_tail(const float* __restrict__ r,
                                                int ka, int kb, int K) {
    v16bf t;
    #pragma unroll
    for (int j = 0; j < 8; ++j) {
        int k1 = ka + j, k2 = kb + j;
        t[j]     = (__bf16)(k1 < K ? r[k1] : 0.f);
        t[8 + j] = (__bf16)(k2 < K ? r[k2] : 0.f);
    }
    return t;
}

// ---------------------------------------------------------------------------
// WMMA GEMM:  C[M,N] = act( A[M,K](lda) * W[N,K]^T + bias ) (+ Cadd)
// One wave computes a 32x64 tile: 2 A-frags x 4 B-frags = 8 WMMAs per K-step,
// each weight fragment reused twice (halves L2 weight traffic on the
// HBM-bound lm_head / readout GEMMs). Branch-free inner loop; K remainder
// (K=48) handled once after the loop via template flag. AVG fuses the
// readout's mean over the 2 sampled layers into the A-operand load.
// act: 0=none 1=relu 2=softplus
// ---------------------------------------------------------------------------
template<bool AVG, bool TAILK>
__global__ void gemm_wmma_bf16(const float* __restrict__ A,
                               const float* __restrict__ A2, int lda,
                               const float* __restrict__ W,
                               const float* __restrict__ bias,
                               const float* __restrict__ Cadd,
                               float* __restrict__ C,
                               int M, int N, int K, int act) {
    const int lane = threadIdx.x & 31;
    const int half = lane >> 4;
    const int l16  = lane & 15;
    const int tM = blockIdx.y, tN = blockIdx.x;

    int mr0 = tM * 32 + l16;        int mc0 = mr0 < M ? mr0 : (M - 1);
    int mr1 = tM * 32 + 16 + l16;   int mc1 = mr1 < M ? mr1 : (M - 1);
    const float* a0  = A + (size_t)mc0 * lda;
    const float* a1  = A + (size_t)mc1 * lda;
    const float* a0b = AVG ? A2 + (size_t)mc0 * lda : nullptr;
    const float* a1b = AVG ? A2 + (size_t)mc1 * lda : nullptr;

    const float* wr[4];
    int ncol[4];
    #pragma unroll
    for (int nt = 0; nt < 4; ++nt) {
        ncol[nt] = tN * 64 + nt * 16 + l16;
        int nc = ncol[nt] < N ? ncol[nt] : (N - 1);
        wr[nt] = W + (size_t)nc * K;
    }

    v8f acc[2][4];
    #pragma unroll
    for (int mf = 0; mf < 2; ++mf)
        #pragma unroll
        for (int nt = 0; nt < 4; ++nt)
            acc[mf][nt] = (v8f){0,0,0,0,0,0,0,0};

    const int Kmain = TAILK ? (K & ~31) : K;
    for (int k0 = 0; k0 < Kmain; k0 += 32) {
        const int ka = k0 + half * 8;
        const int kb = k0 + 16 + half * 8;
        // speculative prefetch of the next weight K-slab (dropped if invalid)
        __builtin_prefetch(wr[0] + k0 + 64, 0, 1);

        v16bf af0 = AVG ? load_frag_avg(a0, a0b, ka, kb) : load_frag(a0, ka, kb);
        v16bf af1 = AVG ? load_frag_avg(a1, a1b, ka, kb) : load_frag(a1, ka, kb);

        #pragma unroll
        for (int nt = 0; nt < 4; ++nt) {
            v16bf bf = load_frag(wr[nt], ka, kb);
            acc[0][nt] = wmma_bf16(af0, bf, acc[0][nt]);
            acc[1][nt] = wmma_bf16(af1, bf, acc[1][nt]);
        }
    }
    if (TAILK && Kmain < K) {
        const int ka = Kmain + half * 8;
        const int kb = Kmain + 16 + half * 8;
        v16bf af0 = load_frag_tail(a0, ka, kb, K);
        v16bf af1 = load_frag_tail(a1, ka, kb, K);
        #pragma unroll
        for (int nt = 0; nt < 4; ++nt) {
            v16bf bf = load_frag_tail(wr[nt], ka, kb, K);
            acc[0][nt] = wmma_bf16(af0, bf, acc[0][nt]);
            acc[1][nt] = wmma_bf16(af1, bf, acc[1][nt]);
        }
    }

    // D layout: VGPR v, lanes 0-15 -> (M = v, N = lane); lanes 16-31 -> (v+8, lane-16)
    #pragma unroll
    for (int mf = 0; mf < 2; ++mf) {
        #pragma unroll
        for (int nt = 0; nt < 4; ++nt) {
            #pragma unroll
            for (int v = 0; v < 8; ++v) {
                int row = tM * 32 + mf * 16 + half * 8 + v;
                int col = ncol[nt];
                if (row < M && col < N) {
                    float val = acc[mf][nt][v];
                    if (bias) val += bias[col];
                    if (act == 1)      val = fmaxf(val, 0.f);
                    else if (act == 2) val = (val > 20.f) ? val
                                             : log1pf(__expf(val));
                    if (Cadd) val += Cadd[(size_t)row * N + col];
                    C[(size_t)row * N + col] = val;
                }
            }
        }
    }
}

// ---------------- embedding gather ----------------
__global__ void embed_k(const int* __restrict__ ids,
                        const float* __restrict__ emb,
                        float* __restrict__ x) {
    int s = blockIdx.x, d = threadIdx.x;       // block = 768
    x[s * DM_ + d] = emb[(size_t)ids[s] * DM_ + d];
}

// ---------------- RMSNorm ----------------
__global__ void rmsnorm_k(const float* __restrict__ x,
                          const float* __restrict__ w,
                          float* __restrict__ o) {
    int s = blockIdx.x, tid = threadIdx.x;     // block = 256, D = 768
    __shared__ float red[256];
    float ss = 0.f;
    for (int d = tid; d < DM_; d += 256) { float v = x[s*DM_+d]; ss += v*v; }
    red[tid] = ss; __syncthreads();
    for (int off = 128; off > 0; off >>= 1) {
        if (tid < off) red[tid] += red[tid + off];
        __syncthreads();
    }
    float inv = rsqrtf(red[0] / (float)DM_ + 1e-5f);
    for (int d = tid; d < DM_; d += 256) o[s*DM_+d] = x[s*DM_+d] * inv * w[d];
}

// ------- causal depthwise conv1d (k=4) + bias + SiLU; reads u-half of xr ----
__global__ void conv_silu_k(const float* __restrict__ xr,
                            const float* __restrict__ cw,
                            const float* __restrict__ cb,
                            float* __restrict__ u) {
    int idx = blockIdx.x * blockDim.x + threadIdx.x;
    if (idx >= S_ * E_) return;
    int t = idx / E_, e = idx % E_;
    float acc = cb[e];
    #pragma unroll
    for (int j = 0; j < 4; ++j) {
        int tt = t - 3 + j;
        if (tt >= 0) acc += xr[(size_t)tt * (2*E_) + e] * cw[e*4 + j];
    }
    u[idx] = acc / (1.f + __expf(-acc));       // SiLU
}

// ---------------------------------------------------------------------------
// SSM scan: each thread owns one (e,n) pair; 16 n-lanes per e-channel live in
// one half-wave (wave32), so the per-timestep  y = sum_n h*C  reduction is 4
// shfl_xor ops. Fuses D-skip and SiLU(res) gate. Optionally stores hs for the
// readout-sampled layers.
// ---------------------------------------------------------------------------
__global__ void scan_k(const float* __restrict__ delta,
                       const float* __restrict__ u,
                       const float* __restrict__ xdbl,
                       const float* __restrict__ xr,
                       const float* __restrict__ Alog,
                       const float* __restrict__ Dv,
                       float* __restrict__ y,
                       float* __restrict__ hs /* null if not sampled */) {
    const int tid = threadIdx.x;               // block = 512 (32 e x 16 n)
    const int n = tid & 15;
    const int e = blockIdx.x * 32 + (tid >> 4);
    const float a = -__expf(Alog[e * NS_ + n]);
    const float dv = Dv[e];
    float h = 0.f;
    for (int t = 0; t < S_; ++t) {
        float d  = delta[(size_t)t * E_ + e];
        float uu = u[(size_t)t * E_ + e];
        float bt = xdbl[(size_t)t * 80 + DR_ + n];
        float ct = xdbl[(size_t)t * 80 + DR_ + NS_ + n];
        h = __expf(d * a) * h + d * bt * uu;
        if (hs) hs[(size_t)t * SD_ + e * NS_ + n] = h;
        float val = h * ct;
        val += __shfl_xor(val, 1, 32);
        val += __shfl_xor(val, 2, 32);
        val += __shfl_xor(val, 4, 32);
        val += __shfl_xor(val, 8, 32);
        if (n == 0) {
            float res = xr[(size_t)t * (2*E_) + E_ + e];
            float yy = val + uu * dv;
            yy *= res / (1.f + __expf(-res));  // * silu(res)
            y[(size_t)t * E_ + e] = yy;
        }
    }
}

// ------- readout attention over ns=2 sampled layers: softmax + weighted v ---
__global__ void readout_attn(const float* __restrict__ qb,
                             const float* __restrict__ kb,
                             const float* __restrict__ vb,
                             float* __restrict__ agg) {
    int s = blockIdx.x, tid = threadIdx.x;     // block = 128
    __shared__ float r0[128], r1[128];
    float q = qb[s * 128 + tid];
    r0[tid] = q * kb[0 * S_ * 128 + s * 128 + tid];
    r1[tid] = q * kb[1 * S_ * 128 + s * 128 + tid];
    __syncthreads();
    for (int off = 64; off > 0; off >>= 1) {
        if (tid < off) { r0[tid] += r0[tid+off]; r1[tid] += r1[tid+off]; }
        __syncthreads();
    }
    const float sc = 0.08838834764831845f;     // 128^-0.5
    float s0 = r0[0] * sc, s1 = r1[0] * sc;
    float m = fmaxf(s0, s1);
    float e0 = __expf(s0 - m), e1 = __expf(s1 - m);
    float inv = 1.f / (e0 + e1);
    float w0 = e0 * inv, w1 = e1 * inv;
    for (int hcol = tid; hcol < HID_; hcol += 128)
        agg[(size_t)s * HID_ + hcol] =
            w0 * vb[0 * S_ * HID_ + (size_t)s * HID_ + hcol] +
            w1 * vb[1 * S_ * HID_ + (size_t)s * HID_ + hcol];
}

// ---------------------------------------------------------------------------
extern "C" void kernel_launch(void* const* d_in, const int* in_sizes, int n_in,
                              void* d_out, int out_size, void* d_ws, size_t ws_size,
                              hipStream_t stream) {
    (void)in_sizes; (void)n_in; (void)out_size; (void)ws_size;

    const int*   ids      = (const int*)  d_in[0];
    const float* emb      = (const float*)d_in[1];
    const float* norm_w   = (const float*)d_in[2];
    const float* in_proj  = (const float*)d_in[3];
    const float* conv_w   = (const float*)d_in[4];
    const float* conv_b   = (const float*)d_in[5];
    const float* x_proj   = (const float*)d_in[6];
    const float* dt_w     = (const float*)d_in[7];
    const float* dt_b     = (const float*)d_in[8];
    const float* A_log    = (const float*)d_in[9];
    const float* D_skip   = (const float*)d_in[10];
    const float* out_proj = (const float*)d_in[11];
    const float* normf_w  = (const float*)d_in[12];
    const float* lm_head  = (const float*)d_in[13];
    const float* q_w = (const float*)d_in[14]; const float* q_b = (const float*)d_in[15];
    const float* k_w = (const float*)d_in[16]; const float* k_b = (const float*)d_in[17];
    const float* v_w = (const float*)d_in[18]; const float* v_b = (const float*)d_in[19];
    const float* o1_w = (const float*)d_in[20]; const float* o1_b = (const float*)d_in[21];
    const float* o2_w = (const float*)d_in[22]; const float* o2_b = (const float*)d_in[23];

    float* out = (float*)d_out;

    // ---- workspace layout (floats) ----
    float* wsp = (float*)d_ws;
    size_t off = 0;
    auto alloc = [&](size_t nf) { float* p = wsp + off; off += nf; return p; };
    float* x     = alloc((size_t)S_ * DM_);
    float* xn    = alloc((size_t)S_ * DM_);
    float* xr    = alloc((size_t)S_ * 2 * E_);
    float* ubuf  = alloc((size_t)S_ * E_);
    float* xdbl  = alloc((size_t)S_ * 80);
    float* delta = alloc((size_t)S_ * E_);
    float* yg    = alloc((size_t)S_ * E_);
    float* hs0   = alloc((size_t)S_ * SD_);
    float* hs4   = alloc((size_t)S_ * SD_);
    float* qb    = alloc((size_t)S_ * 128);
    float* kb    = alloc((size_t)2 * S_ * 128);
    float* vb    = alloc((size_t)2 * S_ * HID_);
    float* agg   = alloc((size_t)S_ * HID_);
    float* hid   = alloc((size_t)S_ * 256);

    auto gemm = [&](const float* A, const float* A2, int lda, const float* W,
                    const float* bias, const float* Cadd, float* C,
                    int M, int N, int K, int act) {
        dim3 grid((N + 63) / 64, (M + 31) / 32);
        if (A2)
            gemm_wmma_bf16<true, false><<<grid, 32, 0, stream>>>(
                A, A2, lda, W, bias, Cadd, C, M, N, K, act);
        else if (K & 31)
            gemm_wmma_bf16<false, true><<<grid, 32, 0, stream>>>(
                A, nullptr, lda, W, bias, Cadd, C, M, N, K, act);
        else
            gemm_wmma_bf16<false, false><<<grid, 32, 0, stream>>>(
                A, nullptr, lda, W, bias, Cadd, C, M, N, K, act);
    };

    // embedding
    embed_k<<<S_, DM_, 0, stream>>>(ids, emb, x);

    for (int i = 0; i < NL_; ++i) {
        rmsnorm_k<<<S_, 256, 0, stream>>>(x, norm_w + (size_t)i * DM_, xn);
        // in_proj: (512,768) x (3072,768)^T
        gemm(xn, nullptr, DM_, in_proj + (size_t)i * 2 * E_ * DM_,
             nullptr, nullptr, xr, S_, 2 * E_, DM_, 0);
        // causal depthwise conv + SiLU
        conv_silu_k<<<(S_ * E_ + 255) / 256, 256, 0, stream>>>(
            xr, conv_w + (size_t)i * E_ * 4, conv_b + (size_t)i * E_, ubuf);
        // x_proj: (512,1536) x (80,1536)^T
        gemm(ubuf, nullptr, E_, x_proj + (size_t)i * 80 * E_,
             nullptr, nullptr, xdbl, S_, 80, E_, 0);
        // delta = softplus(dt @ dt_w^T + dt_b): A = xdbl[:, :48] (lda=80)
        gemm(xdbl, nullptr, 80, dt_w + (size_t)i * E_ * DR_,
             dt_b + (size_t)i * E_, nullptr, delta, S_, E_, DR_, 2);
        // selective scan (+ D-skip + SiLU gate); store hs for layers 0 and 4
        float* hs = (i == 0) ? hs0 : (i == 4) ? hs4 : nullptr;
        scan_k<<<E_ / 32, 512, 0, stream>>>(delta, ubuf, xdbl, xr,
                                            A_log + (size_t)i * E_ * NS_,
                                            D_skip + (size_t)i * E_, yg, hs);
        // out_proj + residual (in-place into x)
        gemm(yg, nullptr, E_, out_proj + (size_t)i * DM_ * E_,
             nullptr, x, x, S_, DM_, E_, 0);
    }

    // main logits: rmsnorm + lm_head (memory-bound: 154 MB fp32 weights,
    // streamed once, converted bf16 in-register)
    rmsnorm_k<<<S_, 256, 0, stream>>>(x, normf_w, xn);
    gemm(xn, nullptr, DM_, lm_head, nullptr, nullptr, out, S_, V_, DM_, 0);

    // readout head
    gemm(hs0, hs4, SD_, q_w, q_b, nullptr, qb, S_, 128, SD_, 1);   // q (mean+relu)
    for (int l = 0; l < 2; ++l) {
        const float* hsl = l ? hs4 : hs0;
        gemm(hsl, nullptr, SD_, k_w, k_b, nullptr, kb + (size_t)l * S_ * 128,
             S_, 128, SD_, 1);                                     // k (relu)
        gemm(hsl, nullptr, SD_, v_w, v_b, nullptr, vb + (size_t)l * S_ * HID_,
             S_, HID_, SD_, 0);                                    // v
    }
    readout_attn<<<S_, 128, 0, stream>>>(qb, kb, vb, agg);
    gemm(agg, nullptr, HID_, o1_w, o1_b, nullptr, hid, S_, 256, HID_, 1);
    gemm(hid, nullptr, 256, o2_w, o2_b, nullptr, out + (size_t)S_ * V_,
         S_, V_, 256, 0);
}
